// GAT_39530878992717
// MI455X (gfx1250) — matrix operand order repaired and checked by reading, not compiled, not run on hardware
//
#include <hip/hip_runtime.h>
#include <math.h>

typedef __attribute__((ext_vector_type(16))) _Float16 v16h;
typedef __attribute__((ext_vector_type(8)))  float    v8f;

#define N_NODES 100000
#define E_EDGES 1600000
#define ET_EDGES (E_EDGES + N_NODES)   // with self-loops
#define KDIM 128                       // inner dim for both layers
#define EPS 1e-5f
#define SLOPE 0.2f

// ---------------- utility ----------------
__global__ void fill_f32(float* __restrict__ p, float v, int n) {
  int i = blockIdx.x * blockDim.x + threadIdx.x;
  int stride = gridDim.x * blockDim.x;
  for (; i < n; i += stride) p[i] = v;
}

// Wt[n*K + k] = (f16) W[k*NC + n]  (transpose + convert, so B-fragments are contiguous)
__global__ void prep_w(const float* __restrict__ W, _Float16* __restrict__ Wt, int NC) {
  int i = blockIdx.x * blockDim.x + threadIdx.x;
  if (i >= NC * KDIM) return;
  int n = i / KDIM, k = i % KDIM;
  Wt[i] = (_Float16)W[k * NC + n];
}

// ---------------- WMMA GEMM: H[N,NC] = X[N,128] * W[128,NC] ----------------
// One wave computes one 16x16 tile; block = NC/16 waves covering one 16-row strip.
// A (16x32 f16): lanes 0-15 row M=lane, K = kb..kb+15, kb = ks*32 (+16 for lanes 16-31).
// B (32x16 f16): lanes 0-15 col N=lane, K = ks*32+e; lanes 16-31 same col, K = ks*32+16+e.
template<int NC>
__global__ __launch_bounds__(32 * (NC / 16)) void gemm_wmma(
    const float* __restrict__ X, const _Float16* __restrict__ Wt,
    float* __restrict__ H) {
  const int lane = threadIdx.x & 31;
  const int wave = threadIdx.x >> 5;
  const int rowbase = blockIdx.x * 16;
  const int colbase = wave * 16;
  const int hi  = (lane >> 4) & 1;
  const int l16 = lane & 15;

  v8f acc = {};
  const float*    xrow = X  + (size_t)(rowbase + l16) * KDIM;
  const _Float16* wcol = Wt + (size_t)(colbase + l16) * KDIM;

#pragma unroll
  for (int ks = 0; ks < KDIM / 32; ++ks) {
    const int kbA = ks * 32 + hi * 16;
    float4 p0 = *(const float4*)(xrow + kbA);
    float4 p1 = *(const float4*)(xrow + kbA + 4);
    float4 q0 = *(const float4*)(xrow + kbA + 8);
    float4 q1 = *(const float4*)(xrow + kbA + 12);
    v16h a;
    a[0]=(_Float16)p0.x;  a[1]=(_Float16)p0.y;  a[2]=(_Float16)p0.z;  a[3]=(_Float16)p0.w;
    a[4]=(_Float16)p1.x;  a[5]=(_Float16)p1.y;  a[6]=(_Float16)p1.z;  a[7]=(_Float16)p1.w;
    a[8]=(_Float16)q0.x;  a[9]=(_Float16)q0.y;  a[10]=(_Float16)q0.z; a[11]=(_Float16)q0.w;
    a[12]=(_Float16)q1.x; a[13]=(_Float16)q1.y; a[14]=(_Float16)q1.z; a[15]=(_Float16)q1.w;

    const int kbB = ks * 32 + hi * 16;
    v16h b = *(const v16h*)(wcol + kbB);   // 16 contiguous f16 = 32B

    acc = __builtin_amdgcn_wmma_f32_16x16x32_f16(false, a, false, b,
                                                 (short)0, acc, false, false);
  }

  // C/D layout: VGPR r -> row M = r (+8 for lanes 16-31), col N = lane&15
  float* orow = H + (size_t)(rowbase + (hi ? 8 : 0)) * NC + colbase + l16;
#pragma unroll
  for (int r = 0; r < 8; ++r) orow[(size_t)r * NC] = acc[r];
}

// ---------------- per-node attention coefficients ----------------
template<int F>
__global__ void alpha_kernel(const float* __restrict__ H,
                             const float* __restrict__ a_s, const float* __restrict__ a_d,
                             float* __restrict__ s, float* __restrict__ d) {
  int wid  = (blockIdx.x * blockDim.x + threadIdx.x) >> 5;
  int lane = threadIdx.x & 31;
  if (wid >= N_NODES) return;
  const float* row = H + (size_t)wid * F;
  float vs = 0.f, vd = 0.f;
#pragma unroll
  for (int j = lane; j < F; j += 32) {
    float h = row[j];
    vs += h * a_s[j];
    vd += h * a_d[j];
  }
#pragma unroll
  for (int off = 16; off > 0; off >>= 1) {
    vs += __shfl_xor(vs, off);
    vd += __shfl_xor(vd, off);
  }
  if (lane == 0) { s[wid] = vs; d[wid] = vd; }
}

// ---------------- edge phase ----------------
__device__ __forceinline__ void atomicMaxF(float* addr, float val) {
  if (val >= 0.f) atomicMax((int*)addr, __float_as_int(val));
  else            atomicMin((unsigned int*)addr, __float_as_uint(val));
}

__device__ __forceinline__ void edge_ends(long e, const int* __restrict__ ei,
                                          int& src, int& dst) {
  if (e < E_EDGES) { src = ei[e]; dst = ei[E_EDGES + e]; }
  else             { src = dst = (int)(e - E_EDGES); }       // self-loop
}

__global__ void edge_logit_max(const int* __restrict__ ei,
                               const float* __restrict__ s, const float* __restrict__ d,
                               float* __restrict__ logit, float* __restrict__ m) {
  long e = (long)blockIdx.x * blockDim.x + threadIdx.x;
  if (e >= ET_EDGES) return;
  int src, dst; edge_ends(e, ei, src, dst);
  float x  = s[src] + d[dst];
  float lr = x > 0.f ? x : SLOPE * x;
  logit[e] = lr;
  atomicMaxF(&m[dst], lr);
}

__global__ void edge_expsum(const int* __restrict__ ei,
                            const float* __restrict__ m,
                            float* __restrict__ wbuf, float* __restrict__ denom) {
  long e = (long)blockIdx.x * blockDim.x + threadIdx.x;
  if (e >= ET_EDGES) return;
  int src, dst; edge_ends(e, ei, src, dst);
  float w = __expf(wbuf[e] - m[dst]);
  wbuf[e] = w;
  atomicAdd(&denom[dst], w);
}

// one wave per edge; lanes split the feature row (float4 / float2 vector traffic)
template<int F>
__global__ void edge_aggregate(const int* __restrict__ ei,
                               const float* __restrict__ wbuf, const float* __restrict__ denom,
                               const float* __restrict__ H, float* __restrict__ agg) {
  long tid = (long)blockIdx.x * blockDim.x + threadIdx.x;
  long e   = tid >> 5;
  int lane = threadIdx.x & 31;
  if (e >= ET_EDGES) return;
  int src, dst; edge_ends(e, ei, src, dst);
  float coef = wbuf[e] / denom[dst];
  if (F == 128) {
    int f0 = lane * 4;
    float4 h = *(const float4*)(H + (size_t)src * F + f0);
    float* o = agg + (size_t)dst * F + f0;
    atomicAdd(o + 0, coef * h.x); atomicAdd(o + 1, coef * h.y);
    atomicAdd(o + 2, coef * h.z); atomicAdd(o + 3, coef * h.w);
  } else {
    int f0 = lane * 2;
    float2 h = *(const float2*)(H + (size_t)src * F + f0);
    float* o = agg + (size_t)dst * F + f0;
    atomicAdd(o + 0, coef * h.x); atomicAdd(o + 1, coef * h.y);
  }
}

// ---------------- BatchNorm ----------------
template<int F>
__global__ void bn_reduce(const float* __restrict__ X,
                          float* __restrict__ sum, float* __restrict__ sumsq) {
  int col = threadIdx.x;                 // blockDim == F
  int rb  = blockIdx.x * 256;
  int re  = rb + 256; if (re > N_NODES) re = N_NODES;
  float s = 0.f, s2 = 0.f;
  for (int r = rb; r < re; ++r) {
    float v = X[(size_t)r * F + col];
    s += v; s2 += v * v;
  }
  atomicAdd(&sum[col], s);
  atomicAdd(&sumsq[col], s2);
}

template<int F>
__global__ void bn_apply_relu(const float* __restrict__ X,
                              const float* __restrict__ sum, const float* __restrict__ sumsq,
                              const float* __restrict__ g, const float* __restrict__ b,
                              float* __restrict__ Y) {
  long i = (long)blockIdx.x * blockDim.x + threadIdx.x;
  long n = (long)N_NODES * F;
  if (i >= n) return;
  int col = (int)(i % F);
  const float inv_n = 1.f / (float)N_NODES;
  float mu  = sum[col] * inv_n;
  float var = sumsq[col] * inv_n - mu * mu;
  float y = (X[i] - mu) * rsqrtf(var + EPS) * g[col] + b[col];
  Y[i] = y > 0.f ? y : 0.f;
}

// ---------------- host-side orchestration ----------------
extern "C" void kernel_launch(void* const* d_in, const int* in_sizes, int n_in,
                              void* d_out, int out_size, void* d_ws, size_t ws_size,
                              hipStream_t stream) {
  const float* x   = (const float*)d_in[0];
  const int*   ei  = (const int*)  d_in[1];
  const float* W0  = (const float*)d_in[2];
  const float* as0 = (const float*)d_in[3];
  const float* ad0 = (const float*)d_in[4];
  const float* g0  = (const float*)d_in[5];
  const float* b0  = (const float*)d_in[6];
  const float* W1  = (const float*)d_in[7];
  const float* as1 = (const float*)d_in[8];
  const float* ad1 = (const float*)d_in[9];
  const float* g1  = (const float*)d_in[10];
  const float* b1  = (const float*)d_in[11];
  float* out = (float*)d_out;

  char* ws = (char*)d_ws;
  size_t off = 0;
  auto alloc = [&](size_t bytes) -> void* {
    void* p = ws + off;
    off = (off + bytes + 255) & ~(size_t)255;
    return p;
  };
  _Float16* Wt0  = (_Float16*)alloc((size_t)128 * 128 * 2);
  _Float16* Wt1  = (_Float16*)alloc((size_t)64  * 128 * 2);
  float* h0      = (float*)alloc((size_t)N_NODES * 128 * 4);   // reused by h1/agg1
  float* agg0    = (float*)alloc((size_t)N_NODES * 128 * 4);
  float* sbuf    = (float*)alloc((size_t)N_NODES * 4);
  float* dbuf    = (float*)alloc((size_t)N_NODES * 4);
  float* mbuf    = (float*)alloc((size_t)N_NODES * 4);
  float* denbuf  = (float*)alloc((size_t)N_NODES * 4);
  float* wbuf    = (float*)alloc((size_t)ET_EDGES * 4);
  float* bnsum   = (float*)alloc(128 * 4);
  float* bnsq    = (float*)alloc(128 * 4);
  float* h1      = h0;                                  // layer-2 features (N x 64)
  float* agg1    = h0 + (size_t)N_NODES * 64;           // layer-2 aggregation (N x 64)

  const int TB = 256;
  const int edge_blocks = (ET_EDGES + TB - 1) / TB;
  const int aggr_blocks = (int)(((long)ET_EDGES * 32 + TB - 1) / TB);
  const int node_wave_blocks = (N_NODES * 32 + TB - 1) / TB;
  const int bn_blocks = (N_NODES + 255) / 256;

  // ---- weight prep (f16 transpose) ----
  prep_w<<<(128 * 128 + TB - 1) / TB, TB, 0, stream>>>(W0, Wt0, 128);
  prep_w<<<(64  * 128 + TB - 1) / TB, TB, 0, stream>>>(W1, Wt1, 64);

  // ================= layer 1 (128 -> 128) =================
  gemm_wmma<128><<<N_NODES / 16, 256, 0, stream>>>(x, Wt0, h0);
  alpha_kernel<128><<<node_wave_blocks, TB, 0, stream>>>(h0, as0, ad0, sbuf, dbuf);

  fill_f32<<<(N_NODES + TB - 1) / TB, TB, 0, stream>>>(mbuf, -INFINITY, N_NODES);
  fill_f32<<<(N_NODES + TB - 1) / TB, TB, 0, stream>>>(denbuf, 0.f, N_NODES);
  fill_f32<<<(N_NODES * 128 + TB - 1) / TB, TB, 0, stream>>>(agg0, 0.f, N_NODES * 128);

  edge_logit_max<<<edge_blocks, TB, 0, stream>>>(ei, sbuf, dbuf, wbuf, mbuf);
  edge_expsum   <<<edge_blocks, TB, 0, stream>>>(ei, mbuf, wbuf, denbuf);
  edge_aggregate<128><<<aggr_blocks, TB, 0, stream>>>(ei, wbuf, denbuf, h0, agg0);

  fill_f32<<<1, 256, 0, stream>>>(bnsum, 0.f, 128);
  fill_f32<<<1, 256, 0, stream>>>(bnsq,  0.f, 128);
  bn_reduce<128><<<bn_blocks, 128, 0, stream>>>(agg0, bnsum, bnsq);
  bn_apply_relu<128><<<(N_NODES * 128 + TB - 1) / TB, TB, 0, stream>>>(
      agg0, bnsum, bnsq, g0, b0, agg0);                // in-place -> layer-2 input

  // ================= layer 2 (128 -> 64) =================
  gemm_wmma<64><<<N_NODES / 16, 128, 0, stream>>>(agg0, Wt1, h1);
  alpha_kernel<64><<<node_wave_blocks, TB, 0, stream>>>(h1, as1, ad1, sbuf, dbuf);

  fill_f32<<<(N_NODES + TB - 1) / TB, TB, 0, stream>>>(mbuf, -INFINITY, N_NODES);
  fill_f32<<<(N_NODES + TB - 1) / TB, TB, 0, stream>>>(denbuf, 0.f, N_NODES);
  fill_f32<<<(N_NODES * 64 + TB - 1) / TB, TB, 0, stream>>>(agg1, 0.f, N_NODES * 64);

  edge_logit_max<<<edge_blocks, TB, 0, stream>>>(ei, sbuf, dbuf, wbuf, mbuf);
  edge_expsum   <<<edge_blocks, TB, 0, stream>>>(ei, mbuf, wbuf, denbuf);
  edge_aggregate<64><<<aggr_blocks, TB, 0, stream>>>(ei, wbuf, denbuf, h1, agg1);

  fill_f32<<<1, 256, 0, stream>>>(bnsum, 0.f, 64);
  fill_f32<<<1, 256, 0, stream>>>(bnsq,  0.f, 64);
  bn_reduce<64><<<bn_blocks, 64, 0, stream>>>(agg1, bnsum, bnsq);
  bn_apply_relu<64><<<(N_NODES * 64 + TB - 1) / TB, TB, 0, stream>>>(
      agg1, bnsum, bnsq, g1, b1, out);
}
